// Gradient_66700842107408
// MI455X (gfx1250) — matrix operand order repaired and checked by reading, not compiled, not run on hardware
//
#include <hip/hip_runtime.h>
#include <hip/hip_bf16.h>
#include <math.h>

// ---------------------------------------------------------------------------
// Forward-mode "jet" propagation of (value, d/dx, d2/dx2, d3/dx3) through a
// tanh MLP 4 -> 512 -> 512 -> 2.  Per hidden unit we carry 35 symmetric
// channels; the 512x512 layer becomes a bf16 WMMA GEMM (M=512, K=512,
// N = 4096 samples * 48 padded channels), tanh chain rule as epilogue.
// This revision: each wave owns 2 j-tiles x 3 channel-tiles = 6 WMMAs per
// 32-deep K-step (shared B fragments, 2x A reuse, 6 independent acc chains
// so loads stagger instead of s_wait_loadcnt 0 before every WMMA), and the
// chain-rule epilogue now uses all 32 lanes.
// ---------------------------------------------------------------------------

typedef __attribute__((ext_vector_type(16))) __bf16 v16bf;
typedef __attribute__((ext_vector_type(8)))  __bf16 v8bf;
typedef __attribute__((ext_vector_type(8)))  float  v8f;

#define BATCH 4096
#define DIN   4
#define HID   512
#define DOUT  2
#define NCH   35     // 1 + 4 + 10 + 20 symmetric jet channels
#define NCHP  48     // padded to 3 WMMA column tiles of 16

// symmetric index tables (constexpr so unrolled loops constant-fold)
constexpr int HP[10][2] = {{0,0},{0,1},{0,2},{0,3},{1,1},{1,2},{1,3},{2,2},{2,3},{3,3}};
constexpr int TP[20][3] = {
  {0,0,0},{0,0,1},{0,0,2},{0,0,3},{0,1,1},{0,1,2},{0,1,3},{0,2,2},{0,2,3},{0,3,3},
  {1,1,1},{1,1,2},{1,1,3},{1,2,2},{1,2,3},{1,3,3},{2,2,2},{2,2,3},{2,3,3},{3,3,3}};

__host__ __device__ constexpr int hidx(int i, int j) {   // requires i <= j
  constexpr int HOFF[4] = {0, 4, 7, 9};
  return HOFF[i] + (j - i);
}

// ---------------------------------------------------------------------------
// Kernel 0: W2 (f32 [i][j]) -> W2T (bf16 [j][i]); j-major so the WMMA A
// fragment reads contiguous K.
// ---------------------------------------------------------------------------
__global__ void prep_w2t(const float* __restrict__ W2, __bf16* __restrict__ W2T) {
  int idx = blockIdx.x * 256 + threadIdx.x;   // HID*HID threads
  int i = idx >> 9, j = idx & (HID - 1);
  W2T[(size_t)j * HID + i] = (__bf16)W2[idx];
}

// ---------------------------------------------------------------------------
// Kernel 1: layer 1 jets.  z_j = W1^T x + b1 has dz/dx_i = W1[i][j] and zero
// higher derivatives, so the tanh chain rule is closed-form.
// Output A1[b][c][i] bf16, c padded to 48 with zeros.
// ---------------------------------------------------------------------------
__global__ void layer1_jet(const float* __restrict__ x, const float* __restrict__ W1,
                           const float* __restrict__ b1, __bf16* __restrict__ A1) {
  int tid = blockIdx.x * 256 + threadIdx.x;   // BATCH*HID threads
  int b = tid >> 9, j = tid & (HID - 1);
  float g[DIN];
  float z = b1[j];
#pragma unroll
  for (int i = 0; i < DIN; ++i) { g[i] = W1[i * HID + j]; z += x[b * DIN + i] * g[i]; }
  float a  = tanhf(z);
  float s1 = 1.f - a * a;                        // tanh'
  float s2 = -2.f * a * s1;                      // tanh''
  float s3 = -2.f * s1 * (1.f - 3.f * a * a);    // tanh'''
  __bf16* out = A1 + (size_t)b * (NCHP * HID) + j;
  out[0] = (__bf16)a;
#pragma unroll
  for (int i = 0; i < 4; ++i) out[(size_t)(1 + i) * HID] = (__bf16)(s1 * g[i]);
#pragma unroll
  for (int p = 0; p < 10; ++p)
    out[(size_t)(5 + p) * HID] = (__bf16)(s2 * g[HP[p][0]] * g[HP[p][1]]);
#pragma unroll
  for (int q = 0; q < 20; ++q)
    out[(size_t)(15 + q) * HID] = (__bf16)(s3 * g[TP[q][0]] * g[TP[q][1]] * g[TP[q][2]]);
#pragma unroll
  for (int c = NCH; c < NCHP; ++c) out[(size_t)c * HID] = (__bf16)0.f;
}

// ---------------------------------------------------------------------------
// Kernel 2: the hot GEMM.  Z[b][j][c] = sum_i W2T[j][i] * A1[b][i][c].
// One wave owns (sample b, 32-row j-pair, all 3 channel tiles):
//   16 K-steps x 6 v_wmma_f32_16x16x32_bf16 (~103 GFLOP total, matrix-core
//   bound; activation traffic ~400 MB @ 23.3 TB/s is negligible).
// Accumulators go through LDS so lanes can run the channel-coupled tanh
// chain rule per (b, j) row, emitting bf16 A2[b][c][j].
// ---------------------------------------------------------------------------
__global__ __launch_bounds__(256) void layer2_wmma(
    const __bf16* __restrict__ W2T, const __bf16* __restrict__ A1,
    const float* __restrict__ b2, __bf16* __restrict__ A2) {
  __shared__ float zlds[8][32][NCHP + 1];      // +1 pad vs bank conflicts (49KB)

  const int wave = threadIdx.x >> 5, lane = threadIdx.x & 31;
  const int gid  = blockIdx.x * 8 + wave;
  const int b    = gid >> 4;                   // 16 j-pairs per sample
  const int jp   = gid & 15;                   // rows [jp*32, jp*32+32)
  const int row  = lane & 15, kh = lane >> 4;

  // A fragments: M=row of W2T j-tile; lane holds contiguous K chunk
  // [kh*16 .. kh*16+15] (matches B's contiguous K order).
  const __bf16* aBase0 = W2T + (size_t)(jp * 32 + row) * HID + kh * 16;
  const __bf16* aBase1 = aBase0 + (size_t)16 * HID;
  // B fragments: N=col (lane&15), K half selected by lane>>4, contiguous K.
  const __bf16* bBase = A1 + (size_t)b * (NCHP * HID) + (size_t)kh * 16;
  const __bf16* bP0 = bBase + (size_t)(0  + row) * HID;
  const __bf16* bP1 = bBase + (size_t)(16 + row) * HID;
  const __bf16* bP2 = bBase + (size_t)(32 + row) * HID;

  v8f acc[2][3] = {};
  for (int k0 = 0; k0 < HID; k0 += 32) {
    __builtin_prefetch(bP0 + k0 + 32, 0, 1);   // global_prefetch_b8
    __builtin_prefetch(bP1 + k0 + 32, 0, 1);
    __builtin_prefetch(bP2 + k0 + 32, 0, 1);
    union { v16bf v; v8bf h[2]; } af0, af1;
    af0.h[0] = *(const v8bf*)(aBase0 + k0);
    af0.h[1] = *(const v8bf*)(aBase0 + k0 + 8);
    af1.h[0] = *(const v8bf*)(aBase1 + k0);
    af1.h[1] = *(const v8bf*)(aBase1 + k0 + 8);
    v16bf bf0 = *(const v16bf*)(bP0 + k0);
    v16bf bf1 = *(const v16bf*)(bP1 + k0);
    v16bf bf2 = *(const v16bf*)(bP2 + k0);
    acc[0][0] = __builtin_amdgcn_wmma_f32_16x16x32_bf16(false, af0.v, false, bf0, (short)0, acc[0][0], false, false);
    acc[0][1] = __builtin_amdgcn_wmma_f32_16x16x32_bf16(false, af0.v, false, bf1, (short)0, acc[0][1], false, false);
    acc[0][2] = __builtin_amdgcn_wmma_f32_16x16x32_bf16(false, af0.v, false, bf2, (short)0, acc[0][2], false, false);
    acc[1][0] = __builtin_amdgcn_wmma_f32_16x16x32_bf16(false, af1.v, false, bf0, (short)0, acc[1][0], false, false);
    acc[1][1] = __builtin_amdgcn_wmma_f32_16x16x32_bf16(false, af1.v, false, bf1, (short)0, acc[1][1], false, false);
    acc[1][2] = __builtin_amdgcn_wmma_f32_16x16x32_bf16(false, af1.v, false, bf2, (short)0, acc[1][2], false, false);
  }

  // C/D layout: VGPR r -> M = r + (lane>>4)*8, N = lane&15.
#pragma unroll
  for (int r = 0; r < 8; ++r) {
    int jr = kh * 8 + r;
#pragma unroll
    for (int t = 0; t < 2; ++t) {
      zlds[wave][t * 16 + jr][ 0 + row] = acc[t][0][r];
      zlds[wave][t * 16 + jr][16 + row] = acc[t][1][r];
      zlds[wave][t * 16 + jr][32 + row] = acc[t][2][r];
    }
  }
  __syncthreads();

  {
    const int j = jp * 32 + lane;              // all 32 lanes: one row each
    const float* zc = zlds[wave][lane];
    float g[4], h[10];
#pragma unroll
    for (int i = 0; i < 4; ++i)  g[i] = zc[1 + i];
#pragma unroll
    for (int p = 0; p < 10; ++p) h[p] = zc[5 + p];
    float v  = zc[0] + b2[j];
    float a  = tanhf(v);
    float s1 = 1.f - a * a;
    float s2 = -2.f * a * s1;
    float s3 = -2.f * s1 * (1.f - 3.f * a * a);
    __bf16* out = A2 + (size_t)b * (NCHP * HID) + j;
    out[0] = (__bf16)a;
#pragma unroll
    for (int i = 0; i < 4; ++i) out[(size_t)(1 + i) * HID] = (__bf16)(s1 * g[i]);
#pragma unroll
    for (int p = 0; p < 10; ++p) {
      int i = HP[p][0], jj = HP[p][1];
      out[(size_t)(5 + p) * HID] = (__bf16)(s2 * g[i] * g[jj] + s1 * h[p]);
    }
#pragma unroll
    for (int q = 0; q < 20; ++q) {
      int i = TP[q][0], jj = TP[q][1], k = TP[q][2];
      float val = s3 * g[i] * g[jj] * g[k]
                + s2 * (h[hidx(i, jj)] * g[k] + h[hidx(i, k)] * g[jj] + h[hidx(jj, k)] * g[i])
                + s1 * zc[15 + q];
      out[(size_t)(15 + q) * HID] = (__bf16)val;
    }
  }
}

// ---------------------------------------------------------------------------
// Kernel 3: output layer (512 -> 2) is too thin for WMMA; one thread per
// (b, o, c) does a 512-MAC f32 dot and scatters its symmetric channel into
// y / dy / ddy / dddy inside d_out.
// ---------------------------------------------------------------------------
__global__ void layer3_scatter(const __bf16* __restrict__ A2, const float* __restrict__ W3,
                               const float* __restrict__ b3, float* __restrict__ out) {
  int tid = blockIdx.x * 256 + threadIdx.x;
  if (tid >= BATCH * DOUT * NCH) return;
  int c = tid % NCH;
  int o = (tid / NCH) % DOUT;
  int b = tid / (NCH * DOUT);

  const __bf16* hp = A2 + (size_t)b * (NCHP * HID) + (size_t)c * HID;
  float acc = 0.f;
  for (int j0 = 0; j0 < HID; j0 += 8) {
    v8bf hv = *(const v8bf*)(hp + j0);
#pragma unroll
    for (int u = 0; u < 8; ++u) acc += (float)hv[u] * W3[(j0 + u) * DOUT + o];
  }

  float* y    = out;                       // [B,2]
  float* dy   = y    + BATCH * DOUT;       // [B,2,4]
  float* ddy  = dy   + BATCH * DOUT * 4;   // [B,2,4,4]
  float* dddy = ddy  + BATCH * DOUT * 16;  // [B,2,4,4,4]
  int bo = b * DOUT + o;
  if (c == 0) {
    y[bo] = acc + b3[o];
  } else if (c < 5) {
    dy[bo * 4 + (c - 1)] = acc;
  } else if (c < 15) {
    int p = c - 5, i = HP[p][0], j = HP[p][1], base = bo * 16;
    ddy[base + i * 4 + j] = acc;
    ddy[base + j * 4 + i] = acc;
  } else {
    int q = c - 15, i = TP[q][0], j = TP[q][1], k = TP[q][2], base = bo * 64;
    dddy[base + i * 16 + j * 4 + k] = acc;
    dddy[base + i * 16 + k * 4 + j] = acc;
    dddy[base + j * 16 + i * 4 + k] = acc;
    dddy[base + j * 16 + k * 4 + i] = acc;
    dddy[base + k * 16 + i * 4 + j] = acc;
    dddy[base + k * 16 + j * 4 + i] = acc;
  }
}

// ---------------------------------------------------------------------------
extern "C" void kernel_launch(void* const* d_in, const int* in_sizes, int n_in,
                              void* d_out, int out_size, void* d_ws, size_t ws_size,
                              hipStream_t stream) {
  (void)in_sizes; (void)n_in; (void)out_size; (void)ws_size;
  const float* x  = (const float*)d_in[0];
  const float* W1 = (const float*)d_in[1];
  const float* b1 = (const float*)d_in[2];
  const float* W2 = (const float*)d_in[3];
  const float* b2 = (const float*)d_in[4];
  const float* W3 = (const float*)d_in[5];
  const float* b3 = (const float*)d_in[6];

  char* ws = (char*)d_ws;
  const size_t jetBytes = (size_t)BATCH * NCHP * HID * sizeof(__bf16); // 192 MiB
  __bf16* W2T = (__bf16*)ws;                               // 512 KiB
  __bf16* A1  = (__bf16*)(ws + (size_t)(1 << 20));
  __bf16* A2  = (__bf16*)(ws + (size_t)(1 << 20) + jetBytes);

  prep_w2t      <<<(HID * HID) / 256,          256, 0, stream>>>(W2, W2T);
  layer1_jet    <<<(BATCH * HID) / 256,        256, 0, stream>>>(x, W1, b1, A1);
  layer2_wmma   <<<(BATCH * 16) / 8,           256, 0, stream>>>(W2T, A1, b2, A2);
  layer3_scatter<<<(BATCH * DOUT * NCH + 255) / 256, 256, 0, stream>>>(A2, W3, b3, (float*)d_out);
}
